// Encoder_17386027614431
// MI455X (gfx1250) — matrix operand order repaired and checked by reading, not compiled, not run on hardware
//
#include <hip/hip_runtime.h>
#include <hip/hip_bf16.h>

typedef __attribute__((ext_vector_type(16))) _Float16 v16h;
typedef __attribute__((ext_vector_type(8)))  float    v8f;

// ---------------- degree / normalization ----------------

__global__ void k_init_deg(float* deg, int n) {
    int i = blockIdx.x * blockDim.x + threadIdx.x;
    if (i < n) deg[i] = 1.0f;               // self-loop contributes 1
}

__global__ void k_accum_deg(const long long* __restrict__ ei, float* deg, int E) {
    int e = blockIdx.x * blockDim.x + threadIdx.x;
    if (e < E) atomicAdd(&deg[(int)ei[E + e]], 1.0f);   // dst row of edge_index
}

__global__ void k_rsqrt(float* deg, int n) {
    int i = blockIdx.x * blockDim.x + threadIdx.x;
    if (i < n) deg[i] = rsqrtf(deg[i]);     // deg >= 1 always
}

// ---------------- B-operand pre-pack ----------------
// Convert W[K,N] (fp32, row-major) into fragment-ready f16 tiles:
// for tile (nt, ks): 32 lanes x 16 halves contiguous.
// Lane l -> column nt*16+(l&15); half i -> k = ks*32 + ((l<16)?0:16) + i.

__global__ void k_pack_b(const float* __restrict__ W, _Float16* __restrict__ Bp,
                         int K, int N) {
    const int KS = K >> 5, NT = N >> 4;
    int idx = blockIdx.x * blockDim.x + threadIdx.x;
    if (idx >= NT * KS * 32) return;
    const int lane = idx & 31;
    const int t    = idx >> 5;              // nt*KS + ks
    const int ks   = t % KS;
    const int nt   = t / KS;
    const int col  = nt * 16 + (lane & 15);
    const int k0   = ks * 32 + ((lane < 16) ? 0 : 16);
    _Float16* out  = Bp + (size_t)idx * 16;
    #pragma unroll
    for (int i = 0; i < 16; ++i)
        out[i] = (_Float16)W[(size_t)(k0 + i) * N + col];
}

// ---------------- WMMA GEMM: C[M,N] = op(A)[M,K] * W[K,N] ----------------
// blockDim = (32,4): 4 waves cover 4 consecutive M-tiles.
// Each wave computes 4 consecutive N-tiles (A fragment reused 4x per k-step).

template <bool RELU>
__global__ void gemm_wmma(const float* __restrict__ A, const _Float16* __restrict__ Bp,
                          float* __restrict__ C, int Mtiles, int K, int N) {
    const int lane = threadIdx.x;                        // 0..31
    const int mt   = blockIdx.x * blockDim.y + threadIdx.y;
    if (mt >= Mtiles) return;                            // whole wave exits pre-WMMA
    const int ntg  = blockIdx.y * 4;                     // first of 4 n-tiles
    const int KS   = K >> 5;

    const int arow    = mt * 16 + (lane & 15);
    const int a_kbase = (lane < 16) ? 0 : 16;            // contiguous k, matches packed B
    const float* arowp = A + (size_t)arow * K + a_kbase;
    const v16h*  bp    = (const v16h*)Bp + ((size_t)ntg * KS * 32 + lane);
    const int    jstr  = KS * 32;                        // v16h stride between n-tiles

    v8f acc0 = {}, acc1 = {}, acc2 = {}, acc3 = {};

    for (int ks = 0; ks < KS; ++ks) {
        // A fragment: halves 0..15 -> k=+0..15
        const float4* ap = (const float4*)(arowp + ks * 32);
        float4 f0 = ap[0], f1 = ap[1], f2 = ap[2], f3 = ap[3];
        float af[16] = {f0.x, f0.y, f0.z, f0.w, f1.x, f1.y, f1.z, f1.w,
                        f2.x, f2.y, f2.z, f2.w, f3.x, f3.y, f3.z, f3.w};
        v16h a;
        #pragma unroll
        for (int i = 0; i < 16; ++i) {
            float v = af[i];
            if (RELU) v = fmaxf(v, 0.0f);
            a[i] = (_Float16)v;
        }
        const v16h* bt = bp + (size_t)ks * 32;
        v16h b0 = bt[0];
        v16h b1 = bt[jstr];
        v16h b2 = bt[2 * jstr];
        v16h b3 = bt[3 * jstr];
        acc0 = __builtin_amdgcn_wmma_f32_16x16x32_f16(false, a, false, b0, (short)0, acc0, false, false);
        acc1 = __builtin_amdgcn_wmma_f32_16x16x32_f16(false, a, false, b1, (short)0, acc1, false, false);
        acc2 = __builtin_amdgcn_wmma_f32_16x16x32_f16(false, a, false, b2, (short)0, acc2, false, false);
        acc3 = __builtin_amdgcn_wmma_f32_16x16x32_f16(false, a, false, b3, (short)0, acc3, false, false);
    }

    // C/D layout: lane = column; VGPR r -> row r (lanes 0-15) / r+8 (lanes 16-31)
    const int rbase = mt * 16 + ((lane < 16) ? 0 : 8);
    const int colx  = lane & 15;
    float* crow = C + (size_t)rbase * N + ntg * 16 + colx;
    #pragma unroll
    for (int r = 0; r < 8; ++r) {
        float* cp = crow + (size_t)r * N;
        cp[0]  = acc0[r];
        cp[16] = acc1[r];
        cp[32] = acc2[r];
        cp[48] = acc3[r];
    }
}

// ---------------- aggregation ----------------

// Out[i,f] = b[f] + T[i,f] * dis[i]^2     (bias + self-loop term)
__global__ void k_seed(const float* __restrict__ T, const float* __restrict__ dis,
                       const float* __restrict__ bias, float* __restrict__ Out,
                       int total, int F) {
    int idx = blockIdx.x * blockDim.x + threadIdx.x;
    if (idx >= total) return;
    int i = idx / F, f = idx - i * F;
    float d = dis[i];
    Out[idx] = bias[f] + T[idx] * d * d;
}

// Out[dst] += T[src] * dis[src]*dis[dst] for every edge; 4 features per thread.
__global__ void k_scatter(const long long* __restrict__ ei, const float* __restrict__ T,
                          const float* __restrict__ dis, float* Out, int E, int F) {
    int idx = blockIdx.x * blockDim.x + threadIdx.x;
    int groups = F >> 2;
    if (idx >= E * groups) return;
    int e = idx / groups;
    int f = (idx - e * groups) << 2;
    int s = (int)ei[e];
    int d = (int)ei[E + e];
    float nrm = dis[s] * dis[d];
    const float4 tv = *(const float4*)(T + (size_t)s * F + f);
    float* od = Out + (size_t)d * F + f;
    atomicAdd(od + 0, tv.x * nrm);
    atomicAdd(od + 1, tv.y * nrm);
    atomicAdd(od + 2, tv.z * nrm);
    atomicAdd(od + 3, tv.w * nrm);
}

// ---------------- launch ----------------

extern "C" void kernel_launch(void* const* d_in, const int* in_sizes, int n_in,
                              void* d_out, int out_size, void* d_ws, size_t ws_size,
                              hipStream_t stream) {
    const float*     x  = (const float*)d_in[0];
    const float*     W1 = (const float*)d_in[1];
    const float*     b1 = (const float*)d_in[2];
    const float*     W2 = (const float*)d_in[3];
    const float*     b2 = (const float*)d_in[4];
    const float*     W3 = (const float*)d_in[5];
    const float*     b3 = (const float*)d_in[6];
    const long long* ei = (const long long*)d_in[7];

    const int Nn = in_sizes[0] / 256;      // 20000 nodes
    const int E  = in_sizes[7] / 2;        // 320000 edges
    float* out = (float*)d_out;

    // workspace layout
    char* ws = (char*)d_ws;
    size_t off = 0;
    float* dis  = (float*)(ws + off); off += ((size_t)Nn * 4 + 255) & ~(size_t)255;
    float* buf0 = (float*)(ws + off); off += (size_t)Nn * 256 * 4;
    float* buf1 = (float*)(ws + off); off += (size_t)Nn * 256 * 4;
    _Float16* Bp = (_Float16*)(ws + off);  // reused per layer, max 256*256*2 B
    (void)ws_size; (void)n_in; (void)out_size;

    const int TPB = 256;
    const int Mtiles = Nn / 16;            // 1250
    const int gx = (Mtiles + 3) / 4;
    dim3 gblk(32, 4);

    // normalization: dis[i] = rsqrt(1 + indeg(i))
    k_init_deg<<<(Nn + TPB - 1) / TPB, TPB, 0, stream>>>(dis, Nn);
    k_accum_deg<<<(E + TPB - 1) / TPB, TPB, 0, stream>>>(ei, dis, E);
    k_rsqrt<<<(Nn + TPB - 1) / TPB, TPB, 0, stream>>>(dis, Nn);

    // ---- layer 1: T1 = x @ W1 ; A1 = scatter(T1) + b1 ----
    {
        int nfr = (256 / 16) * (256 / 32) * 32;  // pack threads
        k_pack_b<<<(nfr + TPB - 1) / TPB, TPB, 0, stream>>>(W1, Bp, 256, 256);
        gemm_wmma<false><<<dim3(gx, 256 / 64), gblk, 0, stream>>>(x, Bp, buf0, Mtiles, 256, 256);
        k_seed<<<((Nn * 256) + TPB - 1) / TPB, TPB, 0, stream>>>(buf0, dis, b1, buf1, Nn * 256, 256);
        k_scatter<<<((E * 64) + TPB - 1) / TPB, TPB, 0, stream>>>(ei, buf0, dis, buf1, E, 256);
    }

    // ---- layer 2: T2 = relu(A1) @ W2 ; A2 = scatter(T2) + b2 ----
    {
        int nfr = (128 / 16) * (256 / 32) * 32;
        k_pack_b<<<(nfr + TPB - 1) / TPB, TPB, 0, stream>>>(W2, Bp, 256, 128);
        gemm_wmma<true><<<dim3(gx, 128 / 64), gblk, 0, stream>>>(buf1, Bp, buf0, Mtiles, 256, 128);
        k_seed<<<((Nn * 128) + TPB - 1) / TPB, TPB, 0, stream>>>(buf0, dis, b2, buf1, Nn * 128, 128);
        k_scatter<<<((E * 32) + TPB - 1) / TPB, TPB, 0, stream>>>(ei, buf0, dis, buf1, E, 128);
    }

    // ---- layer 3: T3 = relu(A2) @ W3 ; out = scatter(T3) + b3 ----
    {
        int nfr = (64 / 16) * (128 / 32) * 32;
        k_pack_b<<<(nfr + TPB - 1) / TPB, TPB, 0, stream>>>(W3, Bp, 128, 64);
        gemm_wmma<true><<<dim3(gx, 64 / 64), gblk, 0, stream>>>(buf1, Bp, buf0, Mtiles, 128, 64);
        k_seed<<<((Nn * 64) + TPB - 1) / TPB, TPB, 0, stream>>>(buf0, dis, b3, out, Nn * 64, 64);
        k_scatter<<<((E * 16) + TPB - 1) / TPB, TPB, 0, stream>>>(ei, buf0, dis, out, E, 64);
    }
}
